// fusion_module_30709016166519
// MI455X (gfx1250) — matrix operand order repaired and checked by reading, not compiled, not run on hardware
//
#include <hip/hip_runtime.h>

// MI455X / gfx1250, wave32. Matrix math via v_wmma_f32_16x16x32_bf16; A/K tile
// staging via the Tensor Data Mover (tensor_load_to_lds + s_wait_tensorcnt).
// Pipeline: fp32->bf16 cvt -> 4 projection GEMMs -> 2 flash-attention passes
// (row-softmax and col-softmax are flash attention with Q/K swapped) ->
// 2 output GEMMs (+bias, fp32). The 268MB sim matrix is never materialized.

typedef __attribute__((ext_vector_type(16))) __bf16 v16bf;
typedef __attribute__((ext_vector_type(8)))  float  v8f;
typedef unsigned int u32x4 __attribute__((ext_vector_type(4)));
typedef int          i32x4 __attribute__((ext_vector_type(4)));
typedef int          i32x8 __attribute__((ext_vector_type(8)));

struct FragBF { union { unsigned int u[8]; v16bf v; }; };

__device__ __forceinline__ unsigned short f32_to_bf16(float f) {
  unsigned int u = __builtin_bit_cast(unsigned int, f);
  unsigned int r = u + 0x7FFFu + ((u >> 16) & 1u);   // round-to-nearest-even
  return (unsigned short)(r >> 16);
}

__device__ __forceinline__ v8f wmma_bf16(const FragBF& a, const FragBF& b, v8f c) {
  return __builtin_amdgcn_wmma_f32_16x16x32_bf16(false, a.v, false, b.v,
                                                 (short)0, c, false, false);
}

// --- Tensor Data Mover: 2D bf16 tile, global -> LDS, with LDS row padding ---
// D# layout per CDNA5 ISA. data_size=2B (code 1).
// pad_interval code: 3 -> 16 dwords, 4 -> 32 dwords; pad_amount code: n -> n+1 dw.
__device__ __forceinline__ void tdm_load_2d_bf16(
    unsigned lds_byte_off, const void* tile_start,
    unsigned tile_w, unsigned tile_h,          // elements, rows
    unsigned stride_elems,                     // tensor dim0 stride (elements)
    unsigned tensor_w, unsigned tensor_h,      // tensor dims (OOB clip)
    unsigned pad_interval_code, unsigned pad_amount_code) {
  unsigned long long ga = (unsigned long long)(size_t)tile_start;
  u32x4 g0;
  g0[0] = 1u;                                              // count=1, user mode
  g0[1] = lds_byte_off;                                    // lds_addr
  g0[2] = (unsigned)(ga & 0xFFFFFFFFu);                    // global_addr[31:0]
  g0[3] = (unsigned)((ga >> 32) & 0x1FFFFFFu) | (2u << 30);// addr[56:32] | type=2
  i32x8 g1;
  g1[0] = (int)((1u << 16) | (1u << 20) |                  // data_size=2B, pad_en
                (pad_interval_code << 22) | (pad_amount_code << 25));
  g1[1] = (int)((tensor_w & 0xFFFFu) << 16);               // dim0[15:0]
  g1[2] = (int)((tensor_w >> 16) | ((tensor_h & 0xFFFFu) << 16));
  g1[3] = (int)((tensor_h >> 16) | (tile_w << 16));        // dim1[31:16]|tile0
  g1[4] = (int)(tile_h & 0xFFFFu);                         // tile1 | tile2=0
  g1[5] = (int)stride_elems;                               // dim0_stride[31:0]
  g1[6] = 0;
  g1[7] = 0;
  i32x4 z4 = {0, 0, 0, 0};
#if defined(__clang_major__) && __clang_major__ >= 23
  i32x8 z8 = {0, 0, 0, 0, 0, 0, 0, 0};
  __builtin_amdgcn_tensor_load_to_lds(g0, g1, z4, z4, z8, 0);
#else
  __builtin_amdgcn_tensor_load_to_lds(g0, g1, z4, z4, 0);
#endif
}

// ---------------------------------------------------------------- convert ---
__global__ void cvt_f32_bf16(const float* __restrict__ src,
                             unsigned short* __restrict__ dst, int n) {
  int i = blockIdx.x * 256 + threadIdx.x;
  if (i < n) dst[i] = f32_to_bf16(src[i]);
}

// ------------------------------------------------------------------- GEMM ---
// C[M,512] = A[M,512](bf16) * W[512,512](bf16); OUTF: fp32+bias, else bf16.
// WG: 256 thr = 8 waves as 4(M)x2(N); 128x128 tile; k-step 32.
// A tile (128x32, row stride 17 dw) streamed by TDM; W tile transposed manually.
template <bool OUTF>
__global__ __launch_bounds__(256) void gemm_wmma(
    const unsigned short* __restrict__ A,
    const unsigned short* __restrict__ W,
    const float* __restrict__ bias,
    unsigned short* __restrict__ outH,
    float* __restrict__ outF,
    int M) {
  constexpr int K = 512, N = 512;
  __shared__ unsigned int shm[128 * 17 * 2];
  unsigned int* ldsA = shm;                    // LDS byte offset 0 (TDM dest)
  unsigned int* ldsB = shm + 128 * 17;         // [n][k] transposed W tile
  unsigned short* ldsBh = (unsigned short*)ldsB;

  const int tid  = threadIdx.x;
  const int lane = tid & 31, wave = tid >> 5;
  const int gr = lane >> 4, lr = lane & 15;
  const int wm = wave >> 1, wn = wave & 1;
  const int m0 = blockIdx.x * 128;
  const int n0 = blockIdx.y * 128;

  v8f acc[2][4];
#pragma unroll
  for (int i = 0; i < 2; i++)
#pragma unroll
    for (int j = 0; j < 4; j++)
#pragma unroll
      for (int r = 0; r < 8; r++) acc[i][j][r] = 0.f;

  for (int k0 = 0; k0 < K; k0 += 32) {
    // async DMA: A tile 128x32 bf16 -> ldsA with 17-dw row stride (16dw + 1 pad)
    if (tid < 32)
      tdm_load_2d_bf16(0u, A + (size_t)m0 * K + k0, 32u, 128u,
                       (unsigned)K, (unsigned)K, (unsigned)M, 3u, 0u);
    // W tile transposed into [n][k]; consecutive threads -> consecutive n
#pragma unroll
    for (int i = 0; i < 16; i++) {
      int e = tid + i * 256;
      int n = e & 127, k = e >> 7;
      ldsBh[n * 34 + k] = W[(size_t)(k0 + k) * N + n0 + n];
    }
    if (k0 + 32 < K)
      __builtin_prefetch(W + (size_t)(k0 + 32 + (tid & 31)) * N + n0, 0, 0);
    if (tid < 32) __builtin_amdgcn_s_wait_tensorcnt(0);
    __syncthreads();

    FragBF af[2], bfr[4];
#pragma unroll
    for (int mt = 0; mt < 2; mt++) {
      int base = (wm * 32 + mt * 16 + lr) * 17 + gr * 4;
#pragma unroll
      for (int v = 0; v < 4; v++) {
        af[mt].u[v]     = ldsA[base + v];       // k = gr*8 + 2v (+1)
        af[mt].u[4 + v] = ldsA[base + 8 + v];   // k = 16 + gr*8 + 2v (+1)
      }
    }
#pragma unroll
    for (int nt = 0; nt < 4; nt++) {
      int base = (wn * 64 + nt * 16 + lr) * 17 + gr * 4;
#pragma unroll
      for (int v = 0; v < 8; v++) bfr[nt].u[v] = ldsB[base + v + (v & 4)]; // k = gr*8+2v (v<4), 16+gr*8+2(v-4)
    }
#pragma unroll
    for (int mt = 0; mt < 2; mt++)
#pragma unroll
      for (int nt = 0; nt < 4; nt++)
        acc[mt][nt] = wmma_bf16(af[mt], bfr[nt], acc[mt][nt]);
    __syncthreads();
  }

#pragma unroll
  for (int mt = 0; mt < 2; mt++)
#pragma unroll
    for (int nt = 0; nt < 4; nt++)
#pragma unroll
      for (int r = 0; r < 8; r++) {
        int row = m0 + wm * 32 + mt * 16 + r + gr * 8;   // C-layout row
        int col = n0 + wn * 64 + nt * 16 + lr;           // C-layout col
        size_t idx = (size_t)row * N + col;
        if constexpr (OUTF) outF[idx] = acc[mt][nt][r] + bias[col];
        else                outH[idx] = f32_to_bf16(acc[mt][nt][r]);
      }
}

// -------------------------------------------------------- flash attention ---
// 128 thr = 4 waves; each wave owns 16 queries; 32-key blocks; head dim 64.
// Buffers are merged-head layout [b*2048 + n][512], head h at cols h*64..+63.
// K tile streamed by TDM (34-dw row stride via pad); V transposed manually.
__global__ __launch_bounds__(128) void attn_wmma(
    const unsigned short* __restrict__ Q,
    const unsigned short* __restrict__ Kb,
    const unsigned short* __restrict__ Vb,
    unsigned short* __restrict__ O,
    float scale) {
  constexpr int SEQ = 2048;
  __shared__ unsigned int shm[32 * 34 + 64 * 17 + 4 * 16 * 17];
  unsigned int* ldsK = shm;                       // byte offset 0 (TDM dest)
  unsigned int* ldsV = shm + 32 * 34;             // Vt: [d][key] stride 17 dw
  unsigned int* ldsP = shm + 32 * 34 + 64 * 17;   // per-wave P 16x32 bf16
  unsigned short* ldsVh = (unsigned short*)ldsV;

  const int tid  = threadIdx.x;
  const int lane = tid & 31, wave = tid >> 5;
  const int gr = lane >> 4, lr = lane & 15;

  const int qblk = blockIdx.x & 31;
  const int h    = (blockIdx.x >> 5) & 7;
  const int b    = blockIdx.x >> 8;
  const int qb   = qblk * 64 + wave * 16;

  const unsigned int* Qd = (const unsigned int*)Q;

  // Q fragments (A-layout 16x32), resident for whole kernel: 2 aligned runs
  FragBF qf[2];
  {
    size_t rowQ = ((size_t)b * SEQ + qb + lr) * 256 + h * 32;  // dword offs
#pragma unroll
    for (int kt = 0; kt < 2; kt++) {
      size_t base = rowQ + kt * 16 + gr * 4;
#pragma unroll
      for (int v = 0; v < 4; v++) {
        qf[kt].u[v]     = Qd[base + v];
        qf[kt].u[4 + v] = Qd[base + 8 + v];
      }
    }
  }

  float m[8], l[8];
  v8f acc[4];
#pragma unroll
  for (int r = 0; r < 8; r++) { m[r] = -1e30f; l[r] = 0.f; }
#pragma unroll
  for (int nt = 0; nt < 4; nt++)
#pragma unroll
    for (int r = 0; r < 8; r++) acc[nt][r] = 0.f;

  unsigned short* pw = (unsigned short*)(ldsP + wave * 272);

  for (int kbase = 0; kbase < SEQ; kbase += 32) {
    // async DMA: K tile 32x64 bf16 -> ldsK with 34-dw row stride (32dw + 2 pad)
    if (tid < 32)
      tdm_load_2d_bf16(0u, Kb + ((size_t)b * SEQ + kbase) * 512 + h * 64,
                       64u, 32u, 512u, 512u, (unsigned)(2 * SEQ), 4u, 1u);
    // V tile transposed: ldsV[d][key]
#pragma unroll
    for (int i = 0; i < 16; i++) {
      int e = tid + i * 128;
      int d = e & 63, key = e >> 6;
      ldsVh[d * 34 + key] = Vb[((size_t)b * SEQ + kbase + key) * 512 + h * 64 + d];
    }
    if (tid < 32) __builtin_amdgcn_s_wait_tensorcnt(0);
    __syncthreads();

    // S = Q * K^T for 32 keys: two 16x16 C-tiles, each 2 WMMA over d
    v8f s_lo, s_hi;
#pragma unroll
    for (int r = 0; r < 8; r++) { s_lo[r] = 0.f; s_hi[r] = 0.f; }
    FragBF kf;
#pragma unroll
    for (int ds = 0; ds < 2; ds++) {               // keys 0..15 (n = lr)
      int base = lr * 34 + ds * 16 + gr * 4;
#pragma unroll
      for (int v = 0; v < 8; v++) kf.u[v] = ldsK[base + v + (v & 4)];
      s_lo = wmma_bf16(qf[ds], kf, s_lo);
    }
#pragma unroll
    for (int ds = 0; ds < 2; ds++) {               // keys 16..31
      int base = (16 + lr) * 34 + ds * 16 + gr * 4;
#pragma unroll
      for (int v = 0; v < 8; v++) kf.u[v] = ldsK[base + v + (v & 4)];
      s_hi = wmma_bf16(qf[ds], kf, s_hi);
    }

    // Online softmax; stats per row (row = r + gr*8, matching C-layout)
#pragma unroll
    for (int r = 0; r < 8; r++) {
      float vmax = fmaxf(s_lo[r], s_hi[r]);
#pragma unroll
      for (int msk = 1; msk < 16; msk <<= 1)
        vmax = fmaxf(vmax, __shfl_xor(vmax, msk, 16));
      float mn  = fmaxf(m[r], vmax * scale);
      float sc  = __expf(m[r] - mn);
      float plo = __expf(s_lo[r] * scale - mn);
      float phi = __expf(s_hi[r] * scale - mn);
      float rs  = plo + phi;
#pragma unroll
      for (int msk = 1; msk < 16; msk <<= 1)
        rs += __shfl_xor(rs, msk, 16);
      m[r] = mn;
      l[r] = l[r] * sc + rs;
#pragma unroll
      for (int nt = 0; nt < 4; nt++) acc[nt][r] *= sc;
      int prow = (r + gr * 8) * 34;
      pw[prow + lr]      = f32_to_bf16(plo);
      pw[prow + 16 + lr] = f32_to_bf16(phi);
    }
    __syncthreads();   // LDS ordering for P round-trip

    // acc += P(16x32) * V(32x64): reload P in A-layout, Vt in B-layout
    FragBF pf, vf;
    {
      int base = wave * 272 + lr * 17 + gr * 4;
#pragma unroll
      for (int v = 0; v < 4; v++) {
        pf.u[v]     = ldsP[base + v];
        pf.u[4 + v] = ldsP[base + 8 + v];
      }
    }
#pragma unroll
    for (int nt = 0; nt < 4; nt++) {
      int base = (nt * 16 + lr) * 17 + gr * 4;
#pragma unroll
      for (int v = 0; v < 8; v++) vf.u[v] = ldsV[base + v + (v & 4)];
      acc[nt] = wmma_bf16(pf, vf, acc[nt]);
    }
    __syncthreads();   // before next tile overwrites ldsK/ldsV
  }

#pragma unroll
  for (int r = 0; r < 8; r++) {
    float inv = 1.0f / l[r];
    int q = qb + r + gr * 8;
    size_t rowO = ((size_t)b * SEQ + q) * 512 + h * 64;
#pragma unroll
    for (int nt = 0; nt < 4; nt++)
      O[rowO + nt * 16 + lr] = f32_to_bf16(acc[nt][r] * inv);
  }
}

// ------------------------------------------------------------------- host ---
extern "C" void kernel_launch(void* const* d_in, const int* in_sizes, int n_in,
                              void* d_out, int out_size, void* d_ws, size_t ws_size,
                              hipStream_t stream) {
  (void)in_sizes; (void)n_in; (void)out_size; (void)ws_size;
  constexpr int Bb = 2, SEQ = 2048, DIM = 512;
  constexpr int ROWS = Bb * SEQ;              // 4096
  constexpr size_t NT = (size_t)ROWS * DIM;   // 2,097,152
  constexpr size_t WSZ = (size_t)DIM * DIM;   // 262,144
  const float SCALE = 0.125f;                 // 64^-0.5

  const float* x      = (const float*)d_in[0];
  const float* ctx    = (const float*)d_in[1];
  const float* w_qk   = (const float*)d_in[2];
  const float* w_cqk  = (const float*)d_in[3];
  const float* w_v    = (const float*)d_in[4];
  const float* w_cv   = (const float*)d_in[5];
  const float* w_out  = (const float*)d_in[6];
  const float* b_out  = (const float*)d_in[7];
  const float* w_cout = (const float*)d_in[8];
  const float* b_cout = (const float*)d_in[9];
  float* out = (float*)d_out;

  unsigned short* ws = (unsigned short*)d_ws;
  unsigned short* xb   = ws + 0 * NT;
  unsigned short* cb   = ws + 1 * NT;
  unsigned short* qk   = ws + 2 * NT;
  unsigned short* cqk  = ws + 3 * NT;
  unsigned short* vv   = ws + 4 * NT;
  unsigned short* cv   = ws + 5 * NT;
  unsigned short* aA   = ws + 6 * NT;
  unsigned short* aB   = ws + 7 * NT;
  unsigned short* wq   = ws + 8 * NT + 0 * WSZ;
  unsigned short* wcq  = ws + 8 * NT + 1 * WSZ;
  unsigned short* wv2  = ws + 8 * NT + 2 * WSZ;
  unsigned short* wcv2 = ws + 8 * NT + 3 * WSZ;
  unsigned short* wo   = ws + 8 * NT + 4 * WSZ;
  unsigned short* wco  = ws + 8 * NT + 5 * WSZ;

  auto cvt = [&](const float* s, unsigned short* d, int n) {
    cvt_f32_bf16<<<(n + 255) / 256, 256, 0, stream>>>(s, d, n);
  };
  cvt(x, xb, (int)NT);
  cvt(ctx, cb, (int)NT);
  cvt(w_qk, wq, (int)WSZ);
  cvt(w_cqk, wcq, (int)WSZ);
  cvt(w_v, wv2, (int)WSZ);
  cvt(w_cv, wcv2, (int)WSZ);
  cvt(w_out, wo, (int)WSZ);
  cvt(w_cout, wco, (int)WSZ);

  dim3 ggrid(ROWS / 128, DIM / 128);
  // projections (bf16 out)
  gemm_wmma<false><<<ggrid, 256, 0, stream>>>(xb, wq,   nullptr, qk,  nullptr, ROWS);
  gemm_wmma<false><<<ggrid, 256, 0, stream>>>(cb, wcq,  nullptr, cqk, nullptr, ROWS);
  gemm_wmma<false><<<ggrid, 256, 0, stream>>>(xb, wv2,  nullptr, vv,  nullptr, ROWS);
  gemm_wmma<false><<<ggrid, 256, 0, stream>>>(cb, wcv2, nullptr, cv,  nullptr, ROWS);

  // out = FlashAttn(Q=qk, K=cqk, V=cv); context_out = FlashAttn(Q=cqk, K=qk, V=v)
  int agrid = Bb * 8 * (SEQ / 64);   // 512 WGs
  attn_wmma<<<agrid, 128, 0, stream>>>(qk,  cqk, cv, aA, SCALE);
  attn_wmma<<<agrid, 128, 0, stream>>>(cqk, qk,  vv, aB, SCALE);

  // output projections (fp32 + bias) into concatenated d_out
  gemm_wmma<true><<<ggrid, 256, 0, stream>>>(aA, wo,  b_out,  nullptr, out,      ROWS);
  gemm_wmma<true><<<ggrid, 256, 0, stream>>>(aB, wco, b_cout, nullptr, out + NT, ROWS);
}
